// _RuntimeLongcatDecoderLayer_78752520339556
// MI455X (gfx1250) — compile-verified, matrix operand order checked by reading, not run on hardware
//
#include <hip/hip_runtime.h>
#include <cstdint>

// ---------------- problem constants (match reference) ----------------
constexpr int TT = 2048;   // tokens
constexpr int HH = 1024;   // hidden
constexpr int II = 768;    // intermediate
constexpr int EE = 16;     // routed experts
constexpr int NEXP = 24;   // E + zero experts
constexpr float SCALEF = 2.5f;

// ---------------- WMMA types / helpers ----------------
typedef __attribute__((ext_vector_type(16))) __bf16 v16bf;
typedef __attribute__((ext_vector_type(8)))  float  v8f;

constexpr int BM = 64;     // token tile
constexpr int BN = 64;     // output-col tile
constexpr int KC = 32;     // K chunk
constexpr int LDP = 40;    // padded LDS row stride (bf16): 80B rows, 16B aligned

__device__ __forceinline__ v8f wmma_bf16(v16bf a, v16bf b, v8f c) {
  return __builtin_amdgcn_wmma_f32_16x16x32_bf16(false, a, false, b, (short)0, c,
                                                 false, false);
}

// 16x32 bf16 fragment from LDS per the documented gfx1250 16-bit A/B layout.
__device__ __forceinline__ v16bf load_frag(const __bf16* base, int lane) {
  const int r  = lane & 15;
  const int kb = (lane >> 4) << 3;   // 0 or 8
  const __bf16* p = base + r * LDP;
  v16bf f;
#pragma unroll
  for (int v = 0; v < 4; ++v) {
    f[2 * v]     = p[kb + 2 * v];
    f[2 * v + 1] = p[kb + 2 * v + 1];
    f[2 * v + 8] = p[16 + kb + 2 * v];
    f[2 * v + 9] = p[16 + kb + 2 * v + 1];
  }
  return f;
}

// CDNA5 async copy: 16B per lane, global -> LDS, tracked by ASYNCcnt.
// LDS operand is the low 32 bits of the generic pointer (flat-aperture rule).
__device__ __forceinline__ void async_ld_b128(void* lds_dst, const void* gsrc) {
  asm volatile("global_load_async_to_lds_b128 %0, %1, off"
               :: "v"((uint32_t)(uintptr_t)lds_dst),
                  "v"((uint64_t)(uintptr_t)gsrc)
               : "memory");
}
// wait thresholds must match the per-stage async issue count (in-order retire)
#define WAIT_ASYNC_6() asm volatile("s_wait_asynccnt 0x6" ::: "memory")
#define WAIT_ASYNC_4() asm volatile("s_wait_asynccnt 0x4" ::: "memory")
#define WAIT_ASYNC_0() asm volatile("s_wait_asynccnt 0x0" ::: "memory")

// ---------------- kernel 1: fp32 -> bf16 bulk convert ----------------
__global__ void cvt_kernel(const float* __restrict__ src, __bf16* __restrict__ dst,
                           int n4) {
  int i = blockIdx.x * blockDim.x + threadIdx.x;
  if (i >= n4) return;
  float4 v = ((const float4*)src)[i];
  union { __bf16 h[4]; uint2 u; } pk;
  pk.h[0] = (__bf16)v.x; pk.h[1] = (__bf16)v.y;
  pk.h[2] = (__bf16)v.z; pk.h[3] = (__bf16)v.w;
  ((uint2*)dst)[i] = pk.u;
}

// ---------------- kernel 2: router (fp32), combine weights ----------------
__global__ void router_kernel(const float* __restrict__ x,
                              const float* __restrict__ wr,
                              const float* __restrict__ bias,
                              float* __restrict__ cw,   // [T, E]
                              float* __restrict__ zw) { // [T]
  const int wave = threadIdx.x >> 5;
  const int lane = threadIdx.x & 31;
  const int t = blockIdx.x * 8 + wave;

  float acc[NEXP];
#pragma unroll
  for (int e = 0; e < NEXP; ++e) acc[e] = 0.f;

  const float* xp = x + (size_t)t * HH;
  for (int j = 0; j < HH; j += 32) {
    float xv = xp[j + lane];
#pragma unroll
    for (int e = 0; e < NEXP; ++e) acc[e] += xv * wr[e * HH + j + lane];
  }
#pragma unroll
  for (int e = 0; e < NEXP; ++e) {
#pragma unroll
    for (int off = 16; off > 0; off >>= 1)
      acc[e] += __shfl_xor(acc[e], off, 32);
  }

  if (lane == 0) {
    float s[NEXP], sel[NEXP];
#pragma unroll
    for (int e = 0; e < NEXP; ++e) {
      s[e]   = 1.f / (1.f + __expf(-acc[e]));
      sel[e] = s[e] + bias[e];
    }
    int ids[4]; float tw[4]; float wsum = 0.f;
#pragma unroll
    for (int k = 0; k < 4; ++k) {
      int best = 0; float bv = -1e30f;
      for (int e = 0; e < NEXP; ++e) {
        bool taken = false;
        for (int q = 0; q < k; ++q) taken |= (ids[q] == e);
        if (!taken && sel[e] > bv) { bv = sel[e]; best = e; }
      }
      ids[k] = best; tw[k] = s[best]; wsum += s[best];
    }
    const float inv = SCALEF / wsum;
    float cwl[EE];
#pragma unroll
    for (int e = 0; e < EE; ++e) cwl[e] = 0.f;
    float z = 0.f;
#pragma unroll
    for (int k = 0; k < 4; ++k) {
      float wk = tw[k] * inv;
      if (ids[k] >= EE) z += wk; else cwl[ids[k]] += wk;
    }
    for (int e = 0; e < EE; ++e) cw[(size_t)t * EE + e] = cwl[e];
    zw[t] = z;
  }
}

// ---------------- kernel 3: gate+up GEMM (bf16, async double-buffered) ------
// grid: (I/64, T/64, E), block 128 (4 waves, 2x2, 32x32 per wave)
__global__ void __launch_bounds__(128)
gateup_kernel(const __bf16* __restrict__ xb,
              const __bf16* __restrict__ wgb,
              const __bf16* __restrict__ wub,
              __bf16* __restrict__ Aws) {   // [E, T, I] bf16
  __shared__ __bf16 Xs[2][BM * LDP];
  __shared__ __bf16 Gs[2][BN * LDP];
  __shared__ __bf16 Us[2][BN * LDP];

  const int e = blockIdx.z;
  const int tBase = blockIdx.y * BM;
  const int iBase = blockIdx.x * BN;
  const int tid  = threadIdx.x;
  const int lane = tid & 31;
  const int wave = tid >> 5;
  const int wm = wave >> 1, wn = wave & 1;

  const __bf16* wgp = wgb + (size_t)e * II * HH;
  const __bf16* wup = wub + (size_t)e * II * HH;

  const v8f vz = 0.f;
  v8f Gacc[2][2] = {{vz, vz}, {vz, vz}};
  v8f Uacc[2][2] = {{vz, vz}, {vz, vz}};

  const int row = tid >> 1;          // 0..63
  const int c0  = (tid & 1) * 16;    // 0 or 16
  const int lo  = row * LDP + c0;    // LDS element offset for this thread

  // issue one stage of async copies (6 x b128 per thread per K step)
  auto stage = [&](int b, int k) {
    const __bf16* xs = xb  + (size_t)(tBase + row) * HH + k + c0;
    const __bf16* gs = wgp + (size_t)(iBase + row) * HH + k + c0;
    const __bf16* us = wup + (size_t)(iBase + row) * HH + k + c0;
    async_ld_b128(&Xs[b][lo],     xs);
    async_ld_b128(&Xs[b][lo + 8], xs + 8);
    async_ld_b128(&Gs[b][lo],     gs);
    async_ld_b128(&Gs[b][lo + 8], gs + 8);
    async_ld_b128(&Us[b][lo],     us);
    async_ld_b128(&Us[b][lo + 8], us + 8);
  };

  stage(0, 0);
  int buf = 0;
  for (int k = 0; k < HH; k += KC) {
    if (k + KC < HH) { stage(buf ^ 1, k + KC); WAIT_ASYNC_6(); }
    else             { WAIT_ASYNC_0(); }
    __syncthreads();

    v16bf a0 = load_frag(&Xs[buf][(wm * 32)      * LDP], lane);
    v16bf a1 = load_frag(&Xs[buf][(wm * 32 + 16) * LDP], lane);
    v16bf g0 = load_frag(&Gs[buf][(wn * 32)      * LDP], lane);
    v16bf g1 = load_frag(&Gs[buf][(wn * 32 + 16) * LDP], lane);
    v16bf u0 = load_frag(&Us[buf][(wn * 32)      * LDP], lane);
    v16bf u1 = load_frag(&Us[buf][(wn * 32 + 16) * LDP], lane);

    Gacc[0][0] = wmma_bf16(a0, g0, Gacc[0][0]);
    Gacc[0][1] = wmma_bf16(a0, g1, Gacc[0][1]);
    Gacc[1][0] = wmma_bf16(a1, g0, Gacc[1][0]);
    Gacc[1][1] = wmma_bf16(a1, g1, Gacc[1][1]);
    Uacc[0][0] = wmma_bf16(a0, u0, Uacc[0][0]);
    Uacc[0][1] = wmma_bf16(a0, u1, Uacc[0][1]);
    Uacc[1][0] = wmma_bf16(a1, u0, Uacc[1][0]);
    Uacc[1][1] = wmma_bf16(a1, u1, Uacc[1][1]);

    __syncthreads();   // protect buf reuse before next stage issue
    buf ^= 1;
  }

  // epilogue: a = silu(g) * u -> bf16 workspace
#pragma unroll
  for (int mi = 0; mi < 2; ++mi)
#pragma unroll
    for (int ni = 0; ni < 2; ++ni)
#pragma unroll
      for (int r = 0; r < 8; ++r) {
        int ml = ((lane >> 4) << 3) + r;
        int t  = tBase + wm * 32 + mi * 16 + ml;
        int ii = iBase + wn * 32 + ni * 16 + (lane & 15);
        float g = Gacc[mi][ni][r];
        float u = Uacc[mi][ni][r];
        float a = g * (1.f / (1.f + __expf(-g))) * u;
        Aws[((size_t)e * TT + t) * II + ii] = (__bf16)a;
      }
}

// ---------------- kernel 4: down GEMM + masked combine + identity -----------
// grid: (H/64, T/64), block 128. Per-expert Yacc, combine weight applied in
// f32 epilogue; block-uniform skip of all-zero experts; async double buffer.
__global__ void __launch_bounds__(128)
down_kernel(const __bf16* __restrict__ Aws,
            const __bf16* __restrict__ wdb,
            const float* __restrict__ cw,
            const float* __restrict__ zw,
            const float* __restrict__ x,
            float* __restrict__ out) {
  __shared__ __bf16 As[2][BM * LDP];
  __shared__ __bf16 Ws[2][BN * LDP];
  __shared__ float  cws[BM];
  __shared__ float  zws[BM];

  const int tBase = blockIdx.y * BM;
  const int hBase = blockIdx.x * BN;
  const int tid  = threadIdx.x;
  const int lane = tid & 31;
  const int wave = tid >> 5;
  const int wm = wave >> 1, wn = wave & 1;
  const int row = tid >> 1;
  const int c0  = (tid & 1) * 16;
  const int lo  = row * LDP + c0;

  if (tid < BM) zws[tid] = zw[tBase + tid];

  const v8f vz = 0.f;
  v8f Racc[2][2] = {{vz, vz}, {vz, vz}};

  for (int e = 0; e < EE; ++e) {
    float cwv = 0.f;
    if (tid < BM) { cwv = cw[(size_t)(tBase + tid) * EE + e]; cws[tid] = cwv; }
    int active = __syncthreads_or(cwv != 0.f);   // block-uniform expert skip
    if (!active) continue;

    const __bf16* Ap  = Aws + (size_t)e * TT * II;
    const __bf16* wdp = wdb + (size_t)e * HH * II;

    auto stage = [&](int b, int k) {
      const __bf16* as = Ap  + (size_t)(tBase + row) * II + k + c0;
      const __bf16* ws = wdp + (size_t)(hBase + row) * II + k + c0;
      async_ld_b128(&As[b][lo],     as);
      async_ld_b128(&As[b][lo + 8], as + 8);
      async_ld_b128(&Ws[b][lo],     ws);
      async_ld_b128(&Ws[b][lo + 8], ws + 8);
    };

    v8f Yacc[2][2] = {{vz, vz}, {vz, vz}};

    stage(0, 0);
    int buf = 0;
    for (int k = 0; k < II; k += KC) {
      // 4 async copies per stage -> wait threshold 4 (in-order completion)
      if (k + KC < II) { stage(buf ^ 1, k + KC); WAIT_ASYNC_4(); }
      else             { WAIT_ASYNC_0(); }
      __syncthreads();

      v16bf a0 = load_frag(&As[buf][(wm * 32)      * LDP], lane);
      v16bf a1 = load_frag(&As[buf][(wm * 32 + 16) * LDP], lane);
      v16bf w0 = load_frag(&Ws[buf][(wn * 32)      * LDP], lane);
      v16bf w1 = load_frag(&Ws[buf][(wn * 32 + 16) * LDP], lane);

      Yacc[0][0] = wmma_bf16(a0, w0, Yacc[0][0]);
      Yacc[0][1] = wmma_bf16(a0, w1, Yacc[0][1]);
      Yacc[1][0] = wmma_bf16(a1, w0, Yacc[1][0]);
      Yacc[1][1] = wmma_bf16(a1, w1, Yacc[1][1]);

      __syncthreads();
      buf ^= 1;
    }

    // fold combine weight (f32) per token row
#pragma unroll
    for (int mi = 0; mi < 2; ++mi)
#pragma unroll
      for (int ni = 0; ni < 2; ++ni)
#pragma unroll
        for (int r = 0; r < 8; ++r) {
          float sc = cws[wm * 32 + mi * 16 + ((lane >> 4) << 3) + r];
          Racc[mi][ni][r] += sc * Yacc[mi][ni][r];
        }
  }

  // epilogue: out = routed + x * zero_w
#pragma unroll
  for (int mi = 0; mi < 2; ++mi)
#pragma unroll
    for (int ni = 0; ni < 2; ++ni)
#pragma unroll
      for (int r = 0; r < 8; ++r) {
        int ml = ((lane >> 4) << 3) + r;
        int tl = wm * 32 + mi * 16 + ml;
        int t  = tBase + tl;
        int h  = hBase + wn * 32 + ni * 16 + (lane & 15);
        float v = Racc[mi][ni][r] + x[(size_t)t * HH + h] * zws[tl];
        out[(size_t)t * HH + h] = v;
      }
}

// ---------------- launcher ----------------
extern "C" void kernel_launch(void* const* d_in, const int* in_sizes, int n_in,
                              void* d_out, int out_size, void* d_ws, size_t ws_size,
                              hipStream_t stream) {
  const float* x    = (const float*)d_in[0];
  const float* wr   = (const float*)d_in[3];
  const float* bias = (const float*)d_in[4];
  const float* wg   = (const float*)d_in[5];
  const float* wu   = (const float*)d_in[6];
  const float* wd   = (const float*)d_in[7];
  float* out = (float*)d_out;

  const size_t nX = (size_t)TT * HH;        // 2.1M
  const size_t nW = (size_t)EE * II * HH;   // 12.6M per weight tensor
  const size_t nA = (size_t)EE * TT * II;   // 25.2M

  char* ws = (char*)d_ws;
  __bf16* xb  = (__bf16*)ws;                         ws += nX * 2;   //  4 MB
  __bf16* Aws = (__bf16*)ws;                         ws += nA * 2;   // 48 MB
  __bf16* wgb = (__bf16*)ws;                         ws += nW * 2;   // 24 MB
  __bf16* wub = (__bf16*)ws;                         ws += nW * 2;   // 24 MB
  __bf16* wdb = (__bf16*)ws;                         ws += nW * 2;   // 24 MB
  float*  cw  = (float*)ws;                          ws += (size_t)TT * EE * 4;
  float*  zw  = (float*)ws;

  // one-time bf16 conversion; converted weights stay L2-resident (72MB << 192MB)
  cvt_kernel<<<(int)((nX / 4 + 255) / 256), 256, 0, stream>>>(x, xb, (int)(nX / 4));
  cvt_kernel<<<(int)((nW / 4 + 255) / 256), 256, 0, stream>>>(wg, wgb, (int)(nW / 4));
  cvt_kernel<<<(int)((nW / 4 + 255) / 256), 256, 0, stream>>>(wu, wub, (int)(nW / 4));
  cvt_kernel<<<(int)((nW / 4 + 255) / 256), 256, 0, stream>>>(wd, wdb, (int)(nW / 4));

  router_kernel<<<TT / 8, 256, 0, stream>>>(x, wr, bias, cw, zw);

  dim3 g2(II / BN, TT / BM, EE);
  gateup_kernel<<<g2, 128, 0, stream>>>(xb, wgb, wub, Aws);

  dim3 g3(HH / BN, TT / BM);
  down_kernel<<<g3, 128, 0, stream>>>(Aws, wdb, cw, zw, x, out);
}
